// VQEmbedding_19679540150538
// MI455X (gfx1250) — matrix operand order repaired and checked by reading, not compiled, not run on hardware
//
#include <hip/hip_runtime.h>
#include <stdint.h>

// Problem constants (match reference): B=8, N=4096, C=256, K=8192
#define Cq   256
#define Kq   8192
#define BNq  32768          // B*N
#define SBS  264            // padded LDS row stride in bf16 elems (132 dwords:
                            // bank group = 4n mod 64 -> conflict-free b128 reads)

typedef __attribute__((ext_vector_type(16))) __bf16   v16bf;
typedef __attribute__((ext_vector_type(8)))  __bf16   v8bf;
typedef __attribute__((ext_vector_type(4)))  __bf16   v4bf;
typedef __attribute__((ext_vector_type(8)))  float    v8f;
typedef __attribute__((ext_vector_type(4)))  float    f32x4;
typedef __attribute__((ext_vector_type(4)))  uint32_t u32x4;
typedef __attribute__((ext_vector_type(4)))  int      i32x4;

#if __has_builtin(__builtin_amdgcn_global_load_async_to_lds_b128) && \
    __has_builtin(__builtin_amdgcn_s_wait_asynccnt)
#define HAVE_ASYNC 1
// builtin expects (int4 addrspace(1)*, int4 addrspace(3)*, imm offset, imm cpol)
#define AS_GLOBAL(p) ((__attribute__((address_space(1))) i32x4*)(p))
#define AS_LDS(p)    ((__attribute__((address_space(3))) i32x4*)(p))
#else
#define HAVE_ASYNC 0
#endif

// ---------------------------------------------------------------------------
// Kernel 1: decompose codebook rows into bf16 hi/lo planes + row squared norms
// ---------------------------------------------------------------------------
__global__ __launch_bounds__(256)
void vq_prep(const float* __restrict__ cb,
             __bf16* __restrict__ chi, __bf16* __restrict__ clo,
             float* __restrict__ csq) {
  const int wave = (blockIdx.x * blockDim.x + threadIdx.x) >> 5;  // row
  const int lane = threadIdx.x & 31;
  if (wave >= Kq) return;
  const float* src = cb + (size_t)wave * Cq;
  float ss = 0.f;
#pragma unroll
  for (int j = 0; j < 2; ++j) {
    f32x4 v = ((const f32x4*)src)[lane + j * 32];
    v4bf hv, lv;
#pragma unroll
    for (int e = 0; e < 4; ++e) {
      float x = v[e];
      __bf16 hi = (__bf16)x;
      __bf16 lo = (__bf16)(x - (float)hi);
      hv[e] = hi; lv[e] = lo;
      ss += x * x;
    }
    ((v4bf*)chi)[(size_t)wave * (Cq / 4) + lane + j * 32] = hv;
    ((v4bf*)clo)[(size_t)wave * (Cq / 4) + lane + j * 32] = lv;
  }
#pragma unroll
  for (int m = 16; m >= 1; m >>= 1) ss += __shfl_xor(ss, m, 32);
  if (lane == 0) csq[wave] = ss;
}

// ---------------------------------------------------------------------------
// Kernel 2: distances + argmin via bf16x3 compensated WMMA GEMM,
// double-buffered codebook tiles staged with async global->LDS copies.
// ---------------------------------------------------------------------------
__global__ __launch_bounds__(256)
void vq_main(const float* __restrict__ X,
             const __bf16* __restrict__ chi, const __bf16* __restrict__ clo,
             const float* __restrict__ csq,
             int* __restrict__ idxO, float* __restrict__ distO) {
  __shared__ __bf16 sBhi[2][16 * SBS];   // 2 x 8.25 KB hi plane ([n][k], padded)
  __shared__ __bf16 sBlo[2][16 * SBS];   // 2 x 8.25 KB lo plane
  __shared__ float  sCsq[2][16];

  const int tid     = threadIdx.x;
  const int lane    = tid & 31;
  const int wv      = tid >> 5;                    // 0..7
  const int rowBase = blockIdx.x * 128 + wv * 16;
  const int mrow    = lane & 15;                   // A-matrix M = lane%16
  const int h       = lane >> 4;                   // half-wave selector

  // Cooperative-copy mapping: thread -> (tile row, 32-byte chunk)
  const int cprow = tid >> 4;                      // 0..15
  const int cpck  = tid & 15;                      // 0..15, 16 bf16 each

  // ---- Load & decompose A fragments (this wave's 16 rows) + |x|^2
  v16bf ahi[8], alo[8];
  float ss = 0.f;
  const float* xrow = X + (size_t)(rowBase + mrow) * Cq;
#pragma unroll
  for (int c = 0; c < 8; ++c) {
    const int base = c * 32;
    const f32x4* p0 = (const f32x4*)(xrow + base + 8 * h);
    const f32x4* p1 = (const f32x4*)(xrow + base + 16 + 8 * h);
    f32x4 f[4] = { p0[0], p0[1], p1[0], p1[1] };
#pragma unroll
    for (int e = 0; e < 16; ++e) {
      float x = f[e >> 2][e & 3];
      __bf16 hi = (__bf16)x;
      __bf16 lo = (__bf16)(x - (float)hi);
      ahi[c][e] = hi; alo[c][e] = lo;
      ss += x * x;
    }
  }
  ss += __shfl_xor(ss, 16, 32);        // lane L holds |x_{L&15}|^2
  float xs[8];
#pragma unroll
  for (int r = 0; r < 8; ++r) xs[r] = __shfl(ss, r + 8 * h, 32);

  float mind[8]; int mini[8];
#pragma unroll
  for (int r = 0; r < 8; ++r) { mind[r] = 3.4e38f; mini[r] = 0; }

  const int n    = mrow;               // B/C/D column = lane%16
  const int koff = h * 16;             // unused; B loads use k offset 8*h per half-wave
  const int NT   = Kq / 16;            // 512 codebook tiles

  float csqReg = 0.f;                  // staged |c|^2 (tid<16)
#if !HAVE_ASYNC
  u32x4 rgH0, rgH1, rgL0, rgL1;        // fallback reg staging
#endif

  // ---- stage(t, buf): start the copy of tile t into LDS buffer buf
  auto stage = [&](int t, int buf) {
    const __bf16* gh = chi + (size_t)t * 16 * Cq + cprow * Cq + cpck * 16;
    const __bf16* gl = clo + (size_t)t * 16 * Cq + cprow * Cq + cpck * 16;
    __bf16* lh = &sBhi[buf][cprow * SBS + cpck * 16];
    __bf16* ll = &sBlo[buf][cprow * SBS + cpck * 16];
#if HAVE_ASYNC
    __builtin_amdgcn_global_load_async_to_lds_b128(AS_GLOBAL(gh),     AS_LDS(lh),     0, 0);
    __builtin_amdgcn_global_load_async_to_lds_b128(AS_GLOBAL(gh + 8), AS_LDS(lh + 8), 0, 0);
    __builtin_amdgcn_global_load_async_to_lds_b128(AS_GLOBAL(gl),     AS_LDS(ll),     0, 0);
    __builtin_amdgcn_global_load_async_to_lds_b128(AS_GLOBAL(gl + 8), AS_LDS(ll + 8), 0, 0);
    (void)lh; (void)ll;
#else
    rgH0 = ((const u32x4*)gh)[0]; rgH1 = ((const u32x4*)gh)[1];
    rgL0 = ((const u32x4*)gl)[0]; rgL1 = ((const u32x4*)gl)[1];
#endif
    if (tid < 16) csqReg = csq[t * 16 + tid];
  };

  // ---- publish(buf): make buffer buf visible (completes the staged copy)
  auto publish = [&](int buf) {
#if !HAVE_ASYNC
    __bf16* lh = &sBhi[buf][cprow * SBS + cpck * 16];
    __bf16* ll = &sBlo[buf][cprow * SBS + cpck * 16];
    ((u32x4*)lh)[0] = rgH0; ((u32x4*)lh)[1] = rgH1;
    ((u32x4*)ll)[0] = rgL0; ((u32x4*)ll)[1] = rgL1;
#endif
    if (tid < 16) sCsq[buf][tid] = csqReg;
#if HAVE_ASYNC
    __builtin_amdgcn_s_wait_asynccnt(0);
#endif
  };

  // prologue: tile 0 into buffer 0
  stage(0, 0);
  publish(0);
  __syncthreads();

  for (int t = 0; t < NT; ++t) {
    const int cur = t & 1;
    if (t + 1 < NT) stage(t + 1, (t + 1) & 1);   // async copy overlaps compute
    if (t + 2 < NT) {                            // pull t+2 toward near caches
      __builtin_prefetch(chi + (size_t)(t + 2) * 16 * Cq + tid * 32, 0, 3);
      __builtin_prefetch(clo + (size_t)(t + 2) * 16 * Cq + tid * 32, 0, 3);
    }

    v8f acc = {};
#pragma unroll
    for (int c = 0; c < 8; ++c) {
      const int base = c * 32;
      const v8bf* pH = (const v8bf*)(&sBhi[cur][n * SBS + base + 8 * h]);
      const v8bf* pL = (const v8bf*)(&sBlo[cur][n * SBS + base + 8 * h]);
      v8bf bh0 = pH[0], bh1 = pH[2];
      v8bf bl0 = pL[0], bl1 = pL[2];
      v16bf bh = __builtin_shufflevector(bh0, bh1, 0,1,2,3,4,5,6,7,8,9,10,11,12,13,14,15);
      v16bf bl = __builtin_shufflevector(bl0, bl1, 0,1,2,3,4,5,6,7,8,9,10,11,12,13,14,15);
      // dot ~= hi*hi + hi*lo + lo*hi (fp32-accurate to ~2^-16)
      acc = __builtin_amdgcn_wmma_f32_16x16x32_bf16(false, ahi[c], false, bh,
                                                    (short)0, acc, false, false);
      acc = __builtin_amdgcn_wmma_f32_16x16x32_bf16(false, ahi[c], false, bl,
                                                    (short)0, acc, false, false);
      acc = __builtin_amdgcn_wmma_f32_16x16x32_bf16(false, alo[c], false, bh,
                                                    (short)0, acc, false, false);
    }

    const float cs = sCsq[cur][n];
    const int   ng = t * 16 + n;
#pragma unroll
    for (int r = 0; r < 8; ++r) {      // C/D layout: VGPR r -> row r + 8*h
      float d = xs[r] + cs - 2.0f * acc[r];
      if (d < mind[r]) { mind[r] = d; mini[r] = ng; }
    }

    if (t + 1 < NT) publish((t + 1) & 1);
    __syncthreads();
  }

  // ---- cross-lane argmin over the 16 column-residues (tie -> lowest index)
#pragma unroll
  for (int r = 0; r < 8; ++r) {
    float d = mind[r]; int i = mini[r];
#pragma unroll
    for (int m = 1; m < 16; m <<= 1) {
      float od = __shfl_xor(d, m, 32);
      int   oi = __shfl_xor(i, m, 32);
      if (od < d || (od == d && oi < i)) { d = od; i = oi; }
    }
    if (mrow == 0) {
      const int row = rowBase + r + 8 * h;
      idxO[row]  = i;
      distO[row] = d;
    }
  }
}

// ---------------------------------------------------------------------------
// Kernel 3: gather winning codebook rows -> codes. One wave per row.
// ---------------------------------------------------------------------------
__global__ __launch_bounds__(256)
void vq_gather(const float* __restrict__ cb,
               const int* __restrict__ idxO,
               float* __restrict__ codes) {
  const int row  = blockIdx.x * 8 + (threadIdx.x >> 5);
  const int lane = threadIdx.x & 31;
  const int id   = idxO[row];
  const f32x4* src = (const f32x4*)(cb + (size_t)id * Cq);
  f32x4* dst       = (f32x4*)(codes + (size_t)row * Cq);
  dst[lane]      = src[lane];
  dst[lane + 32] = src[lane + 32];
}

// ---------------------------------------------------------------------------
extern "C" void kernel_launch(void* const* d_in, const int* in_sizes, int n_in,
                              void* d_out, int out_size, void* d_ws, size_t ws_size,
                              hipStream_t stream) {
  const float* X  = (const float*)d_in[0];   // z_e_x  [B,N,C] f32
  const float* CB = (const float*)d_in[1];   // codebook [K,C] f32

  float* out   = (float*)d_out;
  float* codes = out;                                       // [BN*C] f32
  int*   idxO  = (int*)(out + (size_t)BNq * Cq);            // [BN] i32
  float* distO = out + (size_t)BNq * Cq + BNq;              // [BN] f32

  // workspace: bf16 hi plane (4 MB) + bf16 lo plane (4 MB) + csq (32 KB)
  __bf16* chi = (__bf16*)d_ws;
  __bf16* clo = chi + (size_t)Kq * Cq;
  float*  csq = (float*)(clo + (size_t)Kq * Cq);

  vq_prep  <<<Kq / 8,    256, 0, stream>>>(CB, chi, clo, csq);
  vq_main  <<<BNq / 128, 256, 0, stream>>>(X, chi, clo, csq, idxO, distO);
  vq_gather<<<BNq / 8,   256, 0, stream>>>(CB, idxO, codes);
}
